// InfoNCE_complete_V1_80410377716010
// MI455X (gfx1250) — matrix operand order repaired and checked by reading, not compiled, not run on hardware
//
#include <hip/hip_runtime.h>
#include <hip/hip_bf16.h>
#include <stdint.h>

typedef __attribute__((ext_vector_type(16))) _Float16 v16h;
typedef __attribute__((ext_vector_type(8)))  _Float16 v8h;
typedef __attribute__((ext_vector_type(4)))  _Float16 v4h;
typedef __attribute__((ext_vector_type(8)))  float    v8f;

#define DIMS 128
#define TEMP_INV 10.0f   // 1/0.1
#define MAXLOGIT 10.0f   // cosine <= 1 -> sim/T <= 10

// ---------------------------------------------------------------- kernel 1
// One wave (32 lanes) per row: L2-normalize, emit fp32 and f16 copies.
__global__ __launch_bounds__(256) void k_normalize(
    const float* __restrict__ emb, float* __restrict__ En,
    _Float16* __restrict__ Eh, int N) {
  int wave = threadIdx.x >> 5, lane = threadIdx.x & 31;
  int row = blockIdx.x * 8 + wave;
  if (row >= N) return;
  const float4* src = (const float4*)(emb + (size_t)row * DIMS);
  float4 v = src[lane];
  float ss = v.x * v.x + v.y * v.y + v.z * v.z + v.w * v.w;
  for (int off = 16; off >= 1; off >>= 1) ss += __shfl_xor(ss, off, 32);
  float inv = 1.0f / sqrtf(ss);
  float4 o = make_float4(v.x * inv, v.y * inv, v.z * inv, v.w * inv);
  ((float4*)(En + (size_t)row * DIMS))[lane] = o;
  v4h h; h.x = (_Float16)o.x; h.y = (_Float16)o.y;
  h.z = (_Float16)o.z; h.w = (_Float16)o.w;
  ((v4h*)(Eh + (size_t)row * DIMS))[lane] = h;
}

// ---------------------------------------------------------------- kernel 2
// One wave per anchor: Gumbel-argmax over same-label j != i (deterministic
// hash RNG standing in for jax.random.gumbel(key 42)), then fp32 pos logit.
__device__ __forceinline__ uint32_t hash2(uint32_t i, uint32_t j) {
  uint32_t x = i * 0x85ebca6bu ^ (j + 0x9e3779b9u + (i << 6) + (i >> 2));
  x ^= x >> 16; x *= 0x7feb352du; x ^= x >> 15; x *= 0x846ca68bu; x ^= x >> 16;
  return x;
}

__global__ __launch_bounds__(256) void k_pos_pick(
    const float* __restrict__ En, const int* __restrict__ labels,
    float* __restrict__ pos_logit, int N) {
  int wave = threadIdx.x >> 5, lane = threadIdx.x & 31;
  int i = blockIdx.x * 8 + wave;
  if (i >= N) return;
  int lab = labels[i];
  float bg = -1e30f; int bj = -1;
  for (int j = lane; j < N; j += 32) {
    if (j == i || labels[j] != lab) continue;
    uint32_t h = hash2((uint32_t)i, (uint32_t)j);
    float u = fmaxf((float)(h >> 8) * (1.0f / 16777216.0f), 1e-7f);
    float g = -__logf(-__logf(u));          // gumbel
    if (g > bg || (g == bg && (unsigned)j < (unsigned)bj)) { bg = g; bj = j; }
  }
  for (int off = 16; off >= 1; off >>= 1) {
    float og = __shfl_xor(bg, off, 32);
    int   oj = __shfl_xor(bj, off, 32);
    if (og > bg || (og == bg && (unsigned)oj < (unsigned)bj)) { bg = og; bj = oj; }
  }
  // fp32 dot(E[i], E[bj]) * 1/T
  float4 a = ((const float4*)(En + (size_t)i  * DIMS))[lane];
  float4 b = ((const float4*)(En + (size_t)bj * DIMS))[lane];
  float d = a.x * b.x + a.y * b.y + a.z * b.z + a.w * b.w;
  for (int off = 16; off >= 1; off >>= 1) d += __shfl_xor(d, off, 32);
  if (lane == 0) pos_logit[i] = d * TEMP_INV;
}

// ---------------------------------------------------------------- kernel 3
// Fused sim-GEMM (WMMA f16 -> f32) + masked fixed-max log-sum-exp.
// One wave owns a 32-row anchor block (two 16x16 C tiles), streams all N
// columns in 16-wide tiles; every B tile is reused by 8 WMMAs.
__global__ __launch_bounds__(128) void k_gemm_lse(
    const _Float16* __restrict__ Eh, const int* __restrict__ labels,
    const float* __restrict__ pos_logit, float* __restrict__ loss, int N) {
  int wave = threadIdx.x >> 5, lane = threadIdx.x & 31;
  int Mb = (blockIdx.x * 4 + wave) * 32;
  if (Mb >= N) return;
  int l16 = lane >> 4;        // half-wave id
  int lr  = lane & 15;

  // ---- A tiles: rows Mb..Mb+15 and Mb+16..Mb+31 (ISA 16-bit A 16x32) ----
  // lane group 0 holds K {c*32+0..7, c*32+16..23}; group 1 holds {+8, +24}.
  v16h a0[4], a1[4];
  {
    const _Float16* rp0 = Eh + (size_t)(Mb + lr) * DIMS + l16 * 8;
    const _Float16* rp1 = rp0 + (size_t)16 * DIMS;
#pragma unroll
    for (int c = 0; c < 4; ++c) {
      v8h lo = *(const v8h*)(rp0 + c * 32);
      v8h hi = *(const v8h*)(rp0 + c * 32 + 16);
      v16h av;
      *(v8h*)&av       = lo;
      *((v8h*)&av + 1) = hi;
      a0[c] = av;
      lo = *(const v8h*)(rp1 + c * 32);
      hi = *(const v8h*)(rp1 + c * 32 + 16);
      *(v8h*)&av       = lo;
      *((v8h*)&av + 1) = hi;
      a1[c] = av;
    }
  }
  int rowlab[16];
#pragma unroll
  for (int r = 0; r < 8; ++r) {
    rowlab[r]     = labels[Mb + r + 8 * l16];
    rowlab[r + 8] = labels[Mb + 16 + r + 8 * l16];
  }

  float s[16];
#pragma unroll
  for (int r = 0; r < 16; ++r) s[r] = 0.0f;

  for (int Nb = 0; Nb < N; Nb += 16) {
    // ---- B tile: cols Nb..Nb+15 (ISA 16-bit B 32x16 layout) ----
    // lane group g holds K {c*32 + g*8 .. +7, c*32 + 16 + g*8 .. +7} (same as A).
    const _Float16* cp = Eh + (size_t)(Nb + lr) * DIMS + l16 * 8;
    v8f acc0 = {}, acc1 = {};
#pragma unroll
    for (int c = 0; c < 4; ++c) {
      v8h lo = *(const v8h*)(cp + c * 32);
      v8h hi = *(const v8h*)(cp + c * 32 + 16);
      v16h bv;
      *(v8h*)&bv       = lo;
      *((v8h*)&bv + 1) = hi;
      acc0 = __builtin_amdgcn_wmma_f32_16x16x32_f16(
          false, a0[c], false, bv, (short)0, acc0, false, false);
      acc1 = __builtin_amdgcn_wmma_f32_16x16x32_f16(
          false, a1[c], false, bv, (short)0, acc1, false, false);
    }
    int collab = labels[Nb + lr];  // this lane's column label
#pragma unroll
    for (int r = 0; r < 8; ++r) {
      // element: row = Mb + (tile*16) + r + 8*l16, col = Nb + lr
      float e0 = __expf(acc0[r] * TEMP_INV - MAXLOGIT);
      float e1 = __expf(acc1[r] * TEMP_INV - MAXLOGIT);
      s[r]     += (rowlab[r]     != collab) ? e0 : 0.0f;
      s[r + 8] += (rowlab[r + 8] != collab) ? e1 : 0.0f;
    }
  }

  // reduce each row's partial sums across its 16-lane half-group
#pragma unroll
  for (int r = 0; r < 16; ++r)
    for (int off = 1; off < 16; off <<= 1) s[r] += __shfl_xor(s[r], off, 32);

  if (lr == 0) {   // lane 0 -> rows +0..7 ; lane 16 -> rows +8..15 (per tile)
#pragma unroll
    for (int r = 0; r < 16; ++r) {
      int i = Mb + (r >> 3) * 16 + (r & 7) + 8 * l16;
      float p  = pos_logit[i];
      float m2 = fmaxf(MAXLOGIT, p);
      float tot = s[r] * __expf(MAXLOGIT - m2) + __expf(p - m2);
      loss[i] = m2 + logf(tot) - p;   // lse([pos, negs]) - pos
    }
  }
}

// ---------------------------------------------------------------- kernel 4
__global__ __launch_bounds__(256) void k_reduce_mean(
    const float* __restrict__ loss, float* __restrict__ out, int N) {
  __shared__ float sm[256];
  float acc = 0.0f;
  for (int i = threadIdx.x; i < N; i += 256) acc += loss[i];
  sm[threadIdx.x] = acc;
  __syncthreads();
  for (int s = 128; s > 0; s >>= 1) {
    if (threadIdx.x < s) sm[threadIdx.x] += sm[threadIdx.x + s];
    __syncthreads();
  }
  if (threadIdx.x == 0) out[0] = sm[0] / (float)N;
}

// ---------------------------------------------------------------- launch
extern "C" void kernel_launch(void* const* d_in, const int* in_sizes, int n_in,
                              void* d_out, int out_size, void* d_ws, size_t ws_size,
                              hipStream_t stream) {
  const float* emb    = (const float*)d_in[0];
  const int*   labels = (const int*)d_in[1];
  int N = in_sizes[0] / DIMS;   // 8192

  char* ws = (char*)d_ws;
  float*    En  = (float*)ws;                               // N*128 f32
  _Float16* Eh  = (_Float16*)(ws + (size_t)N * DIMS * 4);   // N*128 f16
  float*    pl  = (float*)(ws + (size_t)N * DIMS * 6);      // N f32
  float*    lo  = pl + N;                                   // N f32
  float*    out = (float*)d_out;

  int rowBlocks = (N + 7) / 8;
  k_normalize<<<rowBlocks, 256, 0, stream>>>(emb, En, Eh, N);
  k_pos_pick <<<rowBlocks, 256, 0, stream>>>(En, labels, pl, N);
  int gemmBlocks = (N / 32 + 3) / 4;
  k_gemm_lse <<<gemmBlocks, 128, 0, stream>>>(Eh, labels, pl, lo, N);
  k_reduce_mean<<<1, 256, 0, stream>>>(lo, out, N);
}
